// Attention_14104672600678
// MI455X (gfx1250) — matrix operand order repaired and checked by reading, not compiled, not run on hardware
//
#include <hip/hip_runtime.h>
#include <hip/hip_bf16.h>
#include <stdint.h>

// Problem constants (fixed by the reference)
#define B_   2
#define L_   2048
#define D_   1024
#define H_   16
#define HD_  64
#define M_TOT (B_*L_)      // 4096 rows
#define SCALE_ 0.125f      // hd^-0.5 = 64^-0.5

typedef __attribute__((ext_vector_type(16))) __bf16 v16bf;
typedef __attribute__((ext_vector_type(8)))  float  v8f;

union FragAB {
  v16bf v;
  unsigned short us[16];
  uint4 q[2];
};

__device__ __forceinline__ unsigned short f2bf(float f) {
  union { float f; unsigned int u; } c; c.f = f;
  unsigned int u = c.u;
  unsigned int r = u + 0x7FFFu + ((u >> 16) & 1u);   // round-to-nearest-even
  if ((u & 0x7F800000u) == 0x7F800000u) r = u;       // keep inf/NaN payload
  return (unsigned short)(r >> 16);
}

// ---------------------------------------------------------------------------
// fp32 -> bf16 elementwise convert, 4-wide (n must be a multiple of 4; all
// tensor sizes here are).
// ---------------------------------------------------------------------------
__global__ void cvt_bf16_kernel(const float* __restrict__ in,
                                unsigned short* __restrict__ out, int n4) {
  int i = blockIdx.x * 256 + threadIdx.x;
  if (i >= n4) return;
  float4 f = ((const float4*)in)[i];
  ushort4 o;
  o.x = f2bf(f.x); o.y = f2bf(f.y); o.z = f2bf(f.z); o.w = f2bf(f.w);
  ((ushort4*)out)[i] = o;
}

// ---------------------------------------------------------------------------
// GEMM: out[M,N] = scale * (A[M,K] @ W[N,K]^T), bf16 inputs, f32 out.
// Block tile 128x128, 8 waves (wave32) in a 4(M)x2(N) grid, wave tile 32x64:
// 2 A-frags x 4 B-frags -> 8 v_wmma_f32_16x16x32_bf16 per K-step.
// LDS staged (2 x 128x32 bf16, padded lead dim), global_prefetch_b8 of the
// next K tile.
// ---------------------------------------------------------------------------
#define LDP 40   // ushorts per LDS row (32 data + 8 pad) -> 80B, 16B aligned
__global__ __launch_bounds__(256, 1)
void gemm_bf16_wmma(const unsigned short* __restrict__ A,
                    const unsigned short* __restrict__ W,
                    float* __restrict__ out,
                    int M, int N, int K, float scale) {
  __shared__ __align__(16) unsigned short As[128 * LDP];
  __shared__ __align__(16) unsigned short Bs[128 * LDP];

  const int tid   = threadIdx.x;
  const int lane  = tid & 31;
  const int wid   = tid >> 5;
  const int laneN = lane & 15;
  const int hi    = lane >> 4;          // 0: lanes 0-15, 1: lanes 16-31
  const int mbase = blockIdx.y * 128;
  const int nbase = blockIdx.x * 128;
  const int wm    = (wid & 3) * 32;     // wave M offset in block tile
  const int wn    = (wid >> 2) * 64;    // wave N offset in block tile
  const int srow  = tid >> 1;           // staging: 128 rows x 2 half-rows
  const int scol  = (tid & 1) * 16;     // 16 bf16 (two uint4) per thread

  v8f acc[2][4] = {};

  for (int k0 = 0; k0 < K; k0 += 32) {
    __syncthreads();
    const unsigned short* ga = A + (size_t)(mbase + srow) * K + k0 + scol;
    const unsigned short* gw = W + (size_t)(nbase + srow) * K + k0 + scol;
    *(uint4*)(As + srow * LDP + scol)     = *(const uint4*)(ga);
    *(uint4*)(As + srow * LDP + scol + 8) = *(const uint4*)(ga + 8);
    *(uint4*)(Bs + srow * LDP + scol)     = *(const uint4*)(gw);
    *(uint4*)(Bs + srow * LDP + scol + 8) = *(const uint4*)(gw + 8);
    if (k0 + 32 < K) {  // global_prefetch_b8 of next K tile
      __builtin_prefetch(ga + 32, 0, 1);
      __builtin_prefetch(gw + 32, 0, 1);
    }
    __syncthreads();

    // A fragments: 16-bit A layout, K interleave {0..7,16..23}/{8..15,24..31}
    FragAB fa[2], fb[4];
#pragma unroll
    for (int mi = 0; mi < 2; ++mi) {
      const unsigned short* ap = As + (wm + mi * 16 + laneN) * LDP;
      fa[mi].q[0] = *(const uint4*)(ap + hi * 8);
      fa[mi].q[1] = *(const uint4*)(ap + 16 + hi * 8);
    }
    // B fragments: same K interleave as A {0..7,16..23}/{8..15,24..31}
#pragma unroll
    for (int ni = 0; ni < 4; ++ni) {
      const unsigned short* bp = Bs + (wn + ni * 16 + laneN) * LDP;
      fb[ni].q[0] = *(const uint4*)(bp + hi * 8);
      fb[ni].q[1] = *(const uint4*)(bp + 16 + hi * 8);
    }
#pragma unroll
    for (int mi = 0; mi < 2; ++mi)
#pragma unroll
      for (int ni = 0; ni < 4; ++ni)
        acc[mi][ni] = __builtin_amdgcn_wmma_f32_16x16x32_bf16(
            false, fa[mi].v, false, fb[ni].v, (short)0, acc[mi][ni],
            false, false);
  }

  // C layout: lanes 0-15 -> M=r, lanes 16-31 -> M=r+8; N = lane&15
#pragma unroll
  for (int mi = 0; mi < 2; ++mi)
#pragma unroll
    for (int ni = 0; ni < 4; ++ni)
#pragma unroll
      for (int r = 0; r < 8; ++r) {
        int m = mbase + wm + mi * 16 + r + 8 * hi;
        out[(size_t)m * N + nbase + wn + ni * 16 + laneN] =
            acc[mi][ni][r] * scale;
      }
}

// ---------------------------------------------------------------------------
// Depthwise conv K=3 along L (+bias), fp32 in -> bf16 out in head-major
// layout [B,H,L,hd] (outRow) and/or transposed [B,H,hd,L] (outT).
// ---------------------------------------------------------------------------
__global__ void dwconv_kernel(const float* __restrict__ in,
                              const float* __restrict__ w,
                              const float* __restrict__ bias,
                              unsigned short* __restrict__ outRow,
                              unsigned short* __restrict__ outT) {
  int idx = blockIdx.x * 256 + threadIdx.x;
  if (idx >= B_ * L_ * D_) return;
  int d   = idx % D_;
  int row = idx / D_;           // row = b*L + l
  int l   = row % L_;
  int b   = row / L_;
  float xm = (l > 0)       ? in[(size_t)(row - 1) * D_ + d] : 0.f;
  float xc =                 in[(size_t)row       * D_ + d];
  float xp = (l < L_ - 1)  ? in[(size_t)(row + 1) * D_ + d] : 0.f;
  float y = w[d * 3 + 0] * xm + w[d * 3 + 1] * xc + w[d * 3 + 2] * xp + bias[d];
  int h = d >> 6, c = d & 63;
  unsigned short bf = f2bf(y);
  if (outRow) outRow[(((size_t)(b * H_ + h)) * L_ + l) * HD_ + c] = bf;
  if (outT)   outT[(((size_t)(b * H_ + h)) * HD_ + c) * L_ + l]   = bf;
}

// ---------------------------------------------------------------------------
// Causal attention core. One block per (b,h, 16-row q tile); 8 waves.
// Full 16x2048 score tile in dynamic LDS (128KB + 4KB ctx, CDNA5 320KB WGP).
// Writes normalized probs to the attn output and bf16 ctx to [B*L, D].
// ---------------------------------------------------------------------------
__global__ __launch_bounds__(256, 1)
void attn_kernel(const unsigned short* __restrict__ qh,   // [B*H, L, hd]
                 const unsigned short* __restrict__ kh,   // [B*H, L, hd] (pre-scaled)
                 const unsigned short* __restrict__ vT,   // [B*H, hd, L]
                 float* __restrict__ attn_out,            // [B*H, L, L]
                 unsigned short* __restrict__ ctx) {      // [B*L, D] bf16
  extern __shared__ __align__(16) char smem[];
  float* sc   = (float*)smem;       // [16][L_]
  float* sctx = sc + 16 * L_;       // [16][HD_]

  const int tid   = threadIdx.x;
  const int lane  = tid & 31;
  const int wid   = tid >> 5;
  const int laneN = lane & 15;
  const int hi    = lane >> 4;
  const int bh    = blockIdx.x / (L_ / 16);
  const int qt    = blockIdx.x % (L_ / 16);
  const int qbase = qt * 16;

  for (int i = tid; i < 16 * HD_; i += 256) sctx[i] = 0.f;

  // ---- Phase 1: scores = q @ k^T with causal mask ----
  FragAB fq0, fq1;
  {
    const unsigned short* qp = qh + ((size_t)bh * L_ + qbase + laneN) * HD_;
    fq0.q[0] = *(const uint4*)(qp + hi * 8);
    fq0.q[1] = *(const uint4*)(qp + 16 + hi * 8);
    fq1.q[0] = *(const uint4*)(qp + 32 + hi * 8);
    fq1.q[1] = *(const uint4*)(qp + 48 + hi * 8);
  }
  const int strip = wid * (L_ / 8);     // 256 columns per wave
  for (int t = 0; t < (L_ / 8) / 16; ++t) {
    const int colbase = strip + t * 16;
    if (colbase > qbase + 15) {         // fully causally masked tile
      for (int r = 0; r < 8; ++r)
        sc[(r + 8 * hi) * L_ + colbase + laneN] = -__builtin_inff();
      continue;
    }
    FragAB fk0, fk1;
    const unsigned short* kp =
        kh + ((size_t)bh * L_ + colbase + laneN) * HD_;
    fk0.q[0] = *(const uint4*)(kp + hi * 8);
    fk0.q[1] = *(const uint4*)(kp + 16 + hi * 8);
    fk1.q[0] = *(const uint4*)(kp + 32 + hi * 8);
    fk1.q[1] = *(const uint4*)(kp + 48 + hi * 8);
    v8f acc = {};
    acc = __builtin_amdgcn_wmma_f32_16x16x32_bf16(false, fq0.v, false, fk0.v,
                                                  (short)0, acc, false, false);
    acc = __builtin_amdgcn_wmma_f32_16x16x32_bf16(false, fq1.v, false, fk1.v,
                                                  (short)0, acc, false, false);
    for (int r = 0; r < 8; ++r) {
      int m   = r + 8 * hi;
      int col = colbase + laneN;
      sc[m * L_ + col] = (col <= qbase + m) ? acc[r] : -__builtin_inff();
    }
  }
  __syncthreads();

  // ---- Phase 2: row softmax (wave32 butterflies) + write attn output ----
  for (int rr = 0; rr < 2; ++rr) {
    const int row = wid * 2 + rr;
    float* srow = sc + row * L_;
    float mx = -__builtin_inff();
    for (int c = lane; c < L_; c += 32) mx = fmaxf(mx, srow[c]);
    for (int off = 16; off > 0; off >>= 1) mx = fmaxf(mx, __shfl_xor(mx, off, 32));
    float s = 0.f;
    for (int c = lane; c < L_; c += 32) {
      float e = __expf(srow[c] - mx);
      srow[c] = e;
      s += e;
    }
    for (int off = 16; off > 0; off >>= 1) s += __shfl_xor(s, off, 32);
    const float inv = 1.f / s;
    float* arow = attn_out + ((size_t)bh * L_ + qbase + row) * L_;
    for (int c = lane; c < L_; c += 32) {
      float p = srow[c] * inv;
      srow[c] = p;
      arow[c] = p;            // coalesced global write of probabilities
    }
  }
  __syncthreads();

  // ---- Phase 3: ctx = attn @ v (partial-K per wave, ds_add_f32 reduce) ----
  v8f acc[4] = {};
  bool any = false;
  for (int ks = 0; ks < 8; ++ks) {
    const int kkbase = strip + ks * 32;
    if (kkbase > qbase + 15) break;     // probs are exactly zero beyond
    any = true;
    FragAB fp;  // A fragment: probs, re-quantized to bf16 (contiguous k = 16*hi + s, same as vT)
    for (int j = 0; j < 8; ++j) {
      fp.us[j]     = f2bf(sc[laneN * L_ + kkbase + hi * 16 + j]);
      fp.us[j + 8] = f2bf(sc[laneN * L_ + kkbase + hi * 16 + 8 + j]);
    }
#pragma unroll
    for (int nt = 0; nt < 4; ++nt) {
      FragAB fv;  // B fragment: vT contiguous along kk
      const unsigned short* vp =
          vT + ((size_t)bh * HD_ + nt * 16 + laneN) * L_ + kkbase + hi * 16;
      fv.q[0] = *(const uint4*)(vp);
      fv.q[1] = *(const uint4*)(vp + 8);
      acc[nt] = __builtin_amdgcn_wmma_f32_16x16x32_bf16(
          false, fp.v, false, fv.v, (short)0, acc[nt], false, false);
    }
  }
  if (any) {
    for (int nt = 0; nt < 4; ++nt)
      for (int r = 0; r < 8; ++r)
        atomicAdd(&sctx[(r + 8 * hi) * HD_ + nt * 16 + laneN], acc[nt][r]);
  }
  __syncthreads();

  // ---- Write ctx as bf16 in [B*L, D] for the output GEMM ----
  const int b = bh / H_, h = bh % H_;
  for (int i = tid; i < 16 * HD_; i += 256) {
    int m = i / HD_, n = i % HD_;
    ctx[((size_t)(b * L_ + qbase + m)) * D_ + h * HD_ + n] = f2bf(sctx[i]);
  }
}

// ---------------------------------------------------------------------------
// Host launcher
// ---------------------------------------------------------------------------
extern "C" void kernel_launch(void* const* d_in, const int* in_sizes, int n_in,
                              void* d_out, int out_size, void* d_ws, size_t ws_size,
                              hipStream_t stream) {
  (void)in_sizes; (void)n_in; (void)out_size; (void)ws_size;
  const float* x   = (const float*)d_in[0];
  const float* wq  = (const float*)d_in[1];
  const float* wk  = (const float*)d_in[2];
  const float* wv  = (const float*)d_in[3];
  const float* wo  = (const float*)d_in[4];
  const float* cqw = (const float*)d_in[5];
  const float* cqb = (const float*)d_in[6];
  const float* ckw = (const float*)d_in[7];
  const float* ckb = (const float*)d_in[8];
  const float* cvw = (const float*)d_in[9];
  const float* cvb = (const float*)d_in[10];

  char* ws = (char*)d_ws;
  size_t off = 0;
  auto alloc = [&](size_t bytes) -> void* {
    void* p = ws + off;
    off += (bytes + 255) & ~size_t(255);
    return p;
  };
  unsigned short* xb   = (unsigned short*)alloc((size_t)M_TOT * D_ * 2);
  unsigned short* wqb  = (unsigned short*)alloc((size_t)D_ * D_ * 2);
  unsigned short* wkb  = (unsigned short*)alloc((size_t)D_ * D_ * 2);
  unsigned short* wvb  = (unsigned short*)alloc((size_t)D_ * D_ * 2);
  unsigned short* wob  = (unsigned short*)alloc((size_t)D_ * D_ * 2);
  float* qproj         = (float*)alloc((size_t)M_TOT * D_ * 4);
  float* kproj         = (float*)alloc((size_t)M_TOT * D_ * 4);
  float* vproj         = (float*)alloc((size_t)M_TOT * D_ * 4);
  unsigned short* qhb  = (unsigned short*)alloc((size_t)B_ * H_ * L_ * HD_ * 2);
  unsigned short* khb  = (unsigned short*)alloc((size_t)B_ * H_ * L_ * HD_ * 2);
  unsigned short* vtb  = (unsigned short*)alloc((size_t)B_ * H_ * L_ * HD_ * 2);
  unsigned short* ctxb = (unsigned short*)alloc((size_t)M_TOT * D_ * 2);

  // 1) fp32 -> bf16 converts (4-wide)
  const int nx = M_TOT * D_;
  const int nw = D_ * D_;
  cvt_bf16_kernel<<<(nx / 4 + 255) / 256, 256, 0, stream>>>(x,  xb,  nx / 4);
  cvt_bf16_kernel<<<(nw / 4 + 255) / 256, 256, 0, stream>>>(wq, wqb, nw / 4);
  cvt_bf16_kernel<<<(nw / 4 + 255) / 256, 256, 0, stream>>>(wk, wkb, nw / 4);
  cvt_bf16_kernel<<<(nw / 4 + 255) / 256, 256, 0, stream>>>(wv, wvb, nw / 4);
  cvt_bf16_kernel<<<(nw / 4 + 255) / 256, 256, 0, stream>>>(wo, wob, nw / 4);

  // 2) QKV projections (k pre-scaled by hd^-0.5, matching reference order)
  dim3 gblk(D_ / 128, M_TOT / 128);
  gemm_bf16_wmma<<<gblk, 256, 0, stream>>>(xb, wqb, qproj, M_TOT, D_, D_, 1.0f);
  gemm_bf16_wmma<<<gblk, 256, 0, stream>>>(xb, wkb, kproj, M_TOT, D_, D_, SCALE_);
  gemm_bf16_wmma<<<gblk, 256, 0, stream>>>(xb, wvb, vproj, M_TOT, D_, D_, 1.0f);

  // 3) depthwise conv + bias; reshuffle to head-major bf16 (v transposed)
  dwconv_kernel<<<(nx + 255) / 256, 256, 0, stream>>>(qproj, cqw, cqb, qhb, nullptr);
  dwconv_kernel<<<(nx + 255) / 256, 256, 0, stream>>>(kproj, ckw, ckb, khb, nullptr);
  dwconv_kernel<<<(nx + 255) / 256, 256, 0, stream>>>(vproj, cvw, cvb, nullptr, vtb);

  // 4) attention (writes attn probs directly into d_out tail)
  float* out  = (float*)d_out;
  float* attn = out + (size_t)M_TOT * D_;
  size_t smem = (size_t)(16 * L_ + 16 * HD_) * sizeof(float);  // 135168 B
  attn_kernel<<<B_ * H_ * (L_ / 16), 256, smem, stream>>>(qhb, khb, vtb, attn, ctxb);

  // 5) output projection into d_out head
  gemm_bf16_wmma<<<gblk, 256, 0, stream>>>(ctxb, wob, out, M_TOT, D_, D_, 1.0f);
}
